// Sboxblock_4638564679873
// MI455X (gfx1250) — matrix operand order repaired and checked by reading, not compiled, not run on hardware
//
#include <hip/hip_runtime.h>
#include <hip/hip_bf16.h>

// ---------------- problem constants ----------------
#define BB    8
#define RR    32
#define BS4   4
#define SHIFT 2
#define CC    128
#define HH    4
#define HD    32
#define NN    64           // tokens per window
#define NW    512          // windows per sample
#define NWIN  (BB * NW)    // 4096 windows total
#define TOK   (BB * RR * RR * RR)  // 262144 rows
#define MLPD  512

typedef __bf16 v16bf __attribute__((ext_vector_type(16)));
typedef float  v8f   __attribute__((ext_vector_type(8)));

union FragU {
  v16bf v;
  unsigned u[8];
};

// round-half-up bf16 (2 VALU)
__device__ __forceinline__ unsigned short f2bf(float x) {
  return (unsigned short)((__float_as_uint(x) + 0x8000u) >> 16);
}
// pack two f32 -> two bf16 in one dword: 2x v_add + 1x v_perm_b32
__device__ __forceinline__ unsigned pack2(float a, float b) {
  return __builtin_amdgcn_perm(__float_as_uint(b) + 0x8000u,
                               __float_as_uint(a) + 0x8000u, 0x07060302u);
}

// A fragment 16x32 bf16 from row-major LDS tile -> 2x ds_load_b128 per lane
__device__ __forceinline__ v16bf loadA128(const unsigned short* base, int row0,
                                          int k0, int stride, int lane) {
  int row = row0 + (lane & 15);
  int hi  = (lane >> 4) & 1;
  const unsigned short* p = base + row * stride + k0 + hi * 16;
  uint4 a = *reinterpret_cast<const uint4*>(p);
  uint4 b = *reinterpret_cast<const uint4*>(p + 8);
  FragU f;
  f.u[0] = a.x; f.u[1] = a.y; f.u[2] = a.z; f.u[3] = a.w;
  f.u[4] = b.x; f.u[5] = b.y; f.u[6] = b.z; f.u[7] = b.w;
  return f.v;
}

// B fragment 32x16 bf16, per-column K-contiguous -> 2x ds_load_b128 per lane
__device__ __forceinline__ v16bf loadB_rm(const unsigned short* base, int col0,
                                          int k0, int rowStride, int lane) {
  int col = col0 + (lane & 15);
  int hi  = (lane >> 4) & 1;
  const unsigned short* p = base + col * rowStride + k0 + hi * 16;
  uint4 a = *reinterpret_cast<const uint4*>(p);
  uint4 b = *reinterpret_cast<const uint4*>(p + 8);
  FragU f;
  f.u[0] = a.x; f.u[1] = a.y; f.u[2] = a.z; f.u[3] = a.w;
  f.u[4] = b.x; f.u[5] = b.y; f.u[6] = b.z; f.u[7] = b.w;
  return f.v;
}

__device__ __forceinline__ v8f wmma_bf16(v16bf a, v16bf b, v8f c) {
  return __builtin_amdgcn_wmma_f32_16x16x32_bf16(false, a, false, b, (short)0, c,
                                                 false, false);
}

// ------- K0a: f32 -> bf16 weight convert with B-fragment swizzle -----------
__global__ void k_cvt_swz(const float* __restrict__ src,
                          unsigned short* __restrict__ dst, int K, int N,
                          int ktMajor) {
  int i = blockIdx.x * 256 + threadIdx.x;
  if (i >= K * N) return;
  int k = i / N, n = i % N;
  int kt = k >> 5, hh = (k >> 4) & 1, kk = k & 15;
  int nt = n >> 4, c = n & 15;
  int ktiles = K >> 5, ntiles = N >> 4;
  int tile = ktMajor ? (kt * ntiles + nt) : (nt * ktiles + kt);
  dst[tile * 512 + (c * 2 + hh) * 16 + kk] = f2bf(src[i]);
}

// ------- K0b: expand relative-position bias to B[h][m][n] (window-invariant)
__global__ void k_bias(const float* __restrict__ bias_table,
                       float* __restrict__ gBias) {
  int i = blockIdx.x * 256 + threadIdx.x;  // 16384 total
  int h = i >> 12, m = (i >> 6) & 63, n = i & 63;
  int ixm = m >> 4, iym = (m >> 2) & 3, izm = m & 3;
  int ixn = n >> 4, iyn = (n >> 2) & 3, izn = n & 3;
  int rpi = (ixm - ixn + 3) * 11 + (iym - iyn + 3) + (izm - izn + 3);
  gBias[i] = bias_table[rpi * HH + h];
}

// ------- K2: fused LN1 + window attention, fully in LDS --------------------
// LDS map (163840 B):
//  A  [0, 98304):      qkv_w swz (p1) | S f32 64K + P bf16 32K (p2) | proj_w 32K (p3)
//  B  [98304, 114688): sX bf16 (p1) | sO bf16 (p3)
//  QK [114688,147456): LN scratch (p0) | sQK 64x256 bf16 row-major (Q 0..127, K 128..255)
//  VT [147456,163840): sVt 128x64 bf16  (Vt[d][token])
#define SM_A  0
#define SM_P  65536
#define SM_B  98304
#define SM_QK 114688
#define SM_VT 147456

__global__ __launch_bounds__(256) void k_attn(
    const float* __restrict__ inp, const unsigned short* __restrict__ qkvw,
    const unsigned short* __restrict__ pw, const float* __restrict__ proj_b,
    const float* __restrict__ gBias, const float* __restrict__ g1,
    const float* __restrict__ b1, float* __restrict__ X) {
  __shared__ __align__(16) char smem[163840];

  int widx = blockIdx.x;
  int t = threadIdx.x, lane = t & 31;
  int wv = __builtin_amdgcn_readfirstlane(t >> 5);  // wave-uniform scalar
  int hi = (lane >> 4) & 1;
  int b = widx >> 9, w = widx & 511;
  int gx = w >> 6, gy = (w >> 3) & 7, gz = w & 7;

  unsigned short* sX  = (unsigned short*)(smem + SM_B);
  unsigned short* sQK = (unsigned short*)(smem + SM_QK);
  unsigned short* sVt = (unsigned short*)(smem + SM_VT);

  // ---- phase 0/1: stage swizzled qkv_w (96 KB) while doing LN1 on the
  //      window's 64 gathered rows directly into sX (no global round-trip) ----
  {
    const uint4* sw = (const uint4*)qkvw;
    uint4* dw = (uint4*)smem;
    for (int i = t; i < (CC * 3 * CC) / 8; i += 256) dw[i] = sw[i];
  }
  {
    float* scr = (float*)(smem + SM_QK);  // dead until QKV outputs are written
    int tok = t >> 2, part = t & 3;
    int ix = tok >> 4, iy = (tok >> 2) & 3, iz = tok & 3;
    int ox = (gx * BS4 + ix + SHIFT) & (RR - 1);
    int oy = (gy * BS4 + iy + SHIFT) & (RR - 1);
    int oz = (gz * BS4 + iz + SHIFT) & (RR - 1);
    long row = (long)b * (RR * RR * RR) + ((ox * RR) + oy) * RR + oz;

    const float4* src4 = (const float4*)(inp + row * CC + part * 32);
    float v[32];
    float s = 0.f, q = 0.f;
#pragma unroll
    for (int j = 0; j < 8; j++) {
      float4 f4 = src4[j];
      v[4 * j] = f4.x; v[4 * j + 1] = f4.y;
      v[4 * j + 2] = f4.z; v[4 * j + 3] = f4.w;
    }
#pragma unroll
    for (int i = 0; i < 32; i++) { s += v[i]; q += v[i] * v[i]; }
    scr[t] = s;
    scr[256 + t] = q;
    __syncthreads();
    if (part == 0) {
      float ts = scr[t] + scr[t + 1] + scr[t + 2] + scr[t + 3];
      float tq = scr[256 + t] + scr[256 + t + 1] + scr[256 + t + 2] + scr[256 + t + 3];
      float m = ts * (1.0f / CC);
      float var = tq * (1.0f / CC) - m * m;
      scr[512 + tok] = m;
      scr[576 + tok] = rsqrtf(var + 1e-5f);
    }
    __syncthreads();
    float m = scr[512 + tok], rs = scr[576 + tok];
    float o[32];
#pragma unroll
    for (int i = 0; i < 32; i++) {
      int c = part * 32 + i;
      o[i] = (v[i] - m) * rs * g1[c] + b1[c];
    }
    uint4* dst4 = (uint4*)(sX + tok * CC + part * 32);
#pragma unroll
    for (int j = 0; j < 4; j++) {
      uint4 pk;
      pk.x = pack2(o[8 * j + 0], o[8 * j + 1]);
      pk.y = pack2(o[8 * j + 2], o[8 * j + 3]);
      pk.z = pack2(o[8 * j + 4], o[8 * j + 5]);
      pk.w = pack2(o[8 * j + 6], o[8 * j + 7]);
      dst4[j] = pk;
    }
  }
  __syncthreads();  // weights + sX ready

  // ---- QKV = xn @ qkv_w : 96 tiles / 8 waves ----
  const unsigned short* sW = (const unsigned short*)smem;
#pragma unroll
  for (int i = 0; i < 12; i++) {
    int tidx = wv * 12 + i;
    int mt = tidx / 24, nt = tidx % 24;
    v8f acc = {0.f, 0.f, 0.f, 0.f, 0.f, 0.f, 0.f, 0.f};
#pragma unroll
    for (int ks = 0; ks < 4; ks++) {
      v16bf a  = loadA128(sX, mt * 16, ks * 32, CC, lane);
      v16bf bb = loadB_rm(sW + (nt * 4 + ks) * 512, 0, 0, 32, lane);
      acc = wmma_bf16(a, bb, acc);
    }
    int m0 = mt * 16 + hi * 8;
    if (nt < 16) {  // Q (scaled) and K -> row-major sQK[token][feat]
      float scale = (nt < 8) ? 0.17677669529663687f : 1.0f;
      int n = nt * 16 + (lane & 15);
#pragma unroll
      for (int r = 0; r < 8; r++) sQK[(m0 + r) * 256 + n] = f2bf(acc[r] * scale);
    } else {        // V -> transposed sVt[d][token], one packed b128 store
      int dg = (nt - 16) * 16 + (lane & 15);
      uint4 pk;
      pk.x = pack2(acc[0], acc[1]);
      pk.y = pack2(acc[2], acc[3]);
      pk.z = pack2(acc[4], acc[5]);
      pk.w = pack2(acc[6], acc[7]);
      *(uint4*)(sVt + dg * 64 + m0) = pk;
    }
  }
  __syncthreads();

  // ---- S = q @ k^T per head : 64 tiles / 8 waves ----
  float* sS = (float*)(smem + SM_A);
#pragma unroll
  for (int i = 0; i < 8; i++) {
    int tidx = wv * 8 + i;
    int h = tidx >> 4, rest = tidx & 15, mt = rest >> 2, nt = rest & 3;
    v16bf a  = loadA128(sQK, mt * 16, h * HD, 256, lane);
    v16bf bb = loadB_rm(sQK, nt * 16, CC + h * HD, 256, lane);
    v8f acc = {0.f, 0.f, 0.f, 0.f, 0.f, 0.f, 0.f, 0.f};
    acc = wmma_bf16(a, bb, acc);
    float* dst = sS + h * (NN * NN);
    int m0 = mt * 16 + hi * 8;
    int n = nt * 16 + (lane & 15);
#pragma unroll
    for (int r = 0; r < 8; r++) dst[(m0 + r) * NN + n] = acc[r];
  }
  __syncthreads();

  // ---- softmax: precomputed bias from global (L2-hot) + border-only mask --
  unsigned short* sP = (unsigned short*)(smem + SM_P);
  {
    int h = t >> 6, m = t & 63;
    const float4* S4 = (const float4*)(sS + h * (NN * NN) + m * NN);
    const float4* B4 = (const float4*)(gBias + h * 4096 + m * 64);
    float vals[NN];
#pragma unroll
    for (int j = 0; j < 16; j++) {
      float4 f4 = S4[j];
      float4 b4 = B4[j];
      vals[4 * j + 0] = f4.x + b4.x;
      vals[4 * j + 1] = f4.y + b4.y;
      vals[4 * j + 2] = f4.z + b4.z;
      vals[4 * j + 3] = f4.w + b4.w;
    }
    if (gx == 7 || gy == 7 || gz == 7) {  // border window: apply shift mask
      int lxm = (gx == 7) ? (1 + ((m >> 5) & 1)) : 0;
      int lym = (gy == 7) ? (1 + ((m >> 3) & 1)) : 0;
      int lzm = (gz == 7) ? (1 + ((m >> 1) & 1)) : 0;
      int labm = lxm * 9 + lym * 3 + lzm;
#pragma unroll
      for (int n = 0; n < NN; n++) {
        int lxn = (gx == 7) ? (1 + ((n >> 5) & 1)) : 0;
        int lyn = (gy == 7) ? (1 + ((n >> 3) & 1)) : 0;
        int lzn = (gz == 7) ? (1 + ((n >> 1) & 1)) : 0;
        if (labm != (lxn * 9 + lyn * 3 + lzn)) vals[n] -= 100.f;
      }
    }
    float mx = -1e30f;
#pragma unroll
    for (int n = 0; n < NN; n++) mx = fmaxf(mx, vals[n]);
    float sum = 0.f;
#pragma unroll
    for (int n = 0; n < NN; n++) {
      vals[n] = __expf(vals[n] - mx);
      sum += vals[n];
    }
    float inv = 1.0f / sum;
    uint4* P4 = (uint4*)(sP + h * (NN * NN) + m * NN);
#pragma unroll
    for (int j = 0; j < 8; j++) {
      uint4 pk;
      pk.x = pack2(vals[8 * j + 0] * inv, vals[8 * j + 1] * inv);
      pk.y = pack2(vals[8 * j + 2] * inv, vals[8 * j + 3] * inv);
      pk.z = pack2(vals[8 * j + 4] * inv, vals[8 * j + 5] * inv);
      pk.w = pack2(vals[8 * j + 6] * inv, vals[8 * j + 7] * inv);
      P4[j] = pk;
    }
  }
  __syncthreads();

  // ---- stage swizzled proj_w (over dead S); O = P @ V : 32 tiles ----
  {
    const uint4* sw = (const uint4*)pw;
    uint4* dw = (uint4*)smem;
    for (int i = t; i < (CC * CC) / 8; i += 256) dw[i] = sw[i];
  }
  unsigned short* sO = (unsigned short*)(smem + SM_B);
#pragma unroll
  for (int i = 0; i < 4; i++) {
    int tidx = wv * 4 + i;
    int h = tidx >> 3, rest = tidx & 7, mt = rest >> 1, nt = rest & 1;
    v8f acc = {0.f, 0.f, 0.f, 0.f, 0.f, 0.f, 0.f, 0.f};
#pragma unroll
    for (int ks = 0; ks < 2; ks++) {
      v16bf a  = loadA128(sP + h * (NN * NN), mt * 16, ks * 32, NN, lane);
      v16bf bb = loadB_rm(sVt + (h * HD) * 64, nt * 16, ks * 32, 64, lane);
      acc = wmma_bf16(a, bb, acc);
    }
    int m0 = mt * 16 + hi * 8;
    int n = h * HD + nt * 16 + (lane & 15);
#pragma unroll
    for (int r = 0; r < 8; r++) sO[(m0 + r) * CC + n] = f2bf(acc[r]);
  }
  __syncthreads();

  // ---- out = O @ proj_w + proj_b; reverse roll + residual scatter ----
  const unsigned short* sPW = (const unsigned short*)smem;
#pragma unroll
  for (int i = 0; i < 4; i++) {
    int tidx = wv * 4 + i;
    int mt = tidx >> 3, nt = tidx & 7;
    v8f acc = {0.f, 0.f, 0.f, 0.f, 0.f, 0.f, 0.f, 0.f};
#pragma unroll
    for (int ks = 0; ks < 4; ks++) {
      v16bf a  = loadA128(sO, mt * 16, ks * 32, CC, lane);
      v16bf bb = loadB_rm(sPW + (nt * 4 + ks) * 512, 0, 0, 32, lane);
      acc = wmma_bf16(a, bb, acc);
    }
    int m0 = mt * 16 + hi * 8;
    int n = nt * 16 + (lane & 15);
    float pb = proj_b[n];
    // hoisted reverse-roll indices: ix == mt constant per tile
    int ox = (gx * BS4 + mt + SHIFT) & (RR - 1);
    long base = (long)b * (RR * RR * RR) + (long)ox * (RR * RR);
    int oyA = (gy * BS4 + hi * 2 + SHIFT) & (RR - 1);
    int oyB = (gy * BS4 + hi * 2 + 1 + SHIFT) & (RR - 1);
    int oz0 = (gz * BS4 + 0 + SHIFT) & (RR - 1);
    int oz1 = (gz * BS4 + 1 + SHIFT) & (RR - 1);
    int oz2 = (gz * BS4 + 2 + SHIFT) & (RR - 1);
    int oz3 = (gz * BS4 + 3 + SHIFT) & (RR - 1);
    int ozt[4] = {oz0, oz1, oz2, oz3};
#pragma unroll
    for (int r = 0; r < 8; r++) {
      int oy = (r < 4) ? oyA : oyB;
      long row = base + oy * RR + ozt[r & 3];
      float val = acc[r] + pb;
      X[row * CC + n] = val * 0.5f + inp[row * CC + n];
    }
  }
}

// ------- K3: LN2 + MLP (4 chunks of 128) + residual ------------------------
__global__ __launch_bounds__(256) void k_mlp(
    const float* __restrict__ X, const float* __restrict__ g2,
    const float* __restrict__ b2, const unsigned short* __restrict__ w1b,
    const float* __restrict__ bb1, const unsigned short* __restrict__ w2b,
    const float* __restrict__ bb2, float* __restrict__ out) {
  __shared__ __align__(16) unsigned short sXn[64 * CC];
  __shared__ __align__(16) unsigned short sW1[CC * CC];
  __shared__ __align__(16) unsigned short sW2[CC * CC];
  __shared__ __align__(16) unsigned short sG[64 * CC];
  __shared__ float s_sum[256], s_sq[256], s_mean[64], s_rstd[64];

  long row0 = (long)blockIdx.x * 64;
  int t = threadIdx.x, lane = t & 31;
  int wv = __builtin_amdgcn_readfirstlane(t >> 5);
  int hi = (lane >> 4) & 1;
  int tok = t >> 2, part = t & 3;

  // LN2
  {
    const float4* src4 = (const float4*)(X + (row0 + tok) * CC + part * 32);
    float v[32];
    float s = 0.f, q = 0.f;
#pragma unroll
    for (int j = 0; j < 8; j++) {
      float4 f4 = src4[j];
      v[4 * j] = f4.x; v[4 * j + 1] = f4.y; v[4 * j + 2] = f4.z; v[4 * j + 3] = f4.w;
    }
#pragma unroll
    for (int i = 0; i < 32; i++) { s += v[i]; q += v[i] * v[i]; }
    s_sum[t] = s;
    s_sq[t] = q;
    __syncthreads();
    if (part == 0) {
      float ts = s_sum[t] + s_sum[t + 1] + s_sum[t + 2] + s_sum[t + 3];
      float tq = s_sq[t] + s_sq[t + 1] + s_sq[t + 2] + s_sq[t + 3];
      float m = ts * (1.0f / CC);
      float var = tq * (1.0f / CC) - m * m;
      s_mean[tok] = m;
      s_rstd[tok] = rsqrtf(var + 1e-5f);
    }
    __syncthreads();
    float m = s_mean[tok], rs = s_rstd[tok];
    float o[32];
#pragma unroll
    for (int i = 0; i < 32; i++) {
      int c = part * 32 + i;
      o[i] = (v[i] - m) * rs * g2[c] + b2[c];
    }
    uint4* dst4 = (uint4*)(sXn + tok * CC + part * 32);
#pragma unroll
    for (int j = 0; j < 4; j++) {
      uint4 pk;
      pk.x = pack2(o[8 * j + 0], o[8 * j + 1]);
      pk.y = pack2(o[8 * j + 2], o[8 * j + 3]);
      pk.z = pack2(o[8 * j + 4], o[8 * j + 5]);
      pk.w = pack2(o[8 * j + 6], o[8 * j + 7]);
      dst4[j] = pk;
    }
  }
  __syncthreads();

  v8f acc[4];
#pragma unroll
  for (int i = 0; i < 4; i++) acc[i] = (v8f){0.f, 0.f, 0.f, 0.f, 0.f, 0.f, 0.f, 0.f};

  for (int cc = 0; cc < 4; cc++) {
    // stage swizzled chunk tiles (both contiguous 32 KB slabs)
    {
      const uint4* s1 = (const uint4*)w1b + cc * 2048;
      uint4* d1 = (uint4*)sW1;
      for (int i = t; i < 2048; i += 256) d1[i] = s1[i];
      const uint4* s2 = (const uint4*)w2b + cc * 2048;
      uint4* d2 = (uint4*)sW2;
      for (int i = t; i < 2048; i += 256) d2[i] = s2[i];
    }
    __syncthreads();

    // g = gelu(xn @ w1_chunk + bb1_chunk)
#pragma unroll
    for (int i2 = 0; i2 < 4; i2++) {
      int tidx = wv * 4 + i2;
      int mt = tidx >> 3, nt = tidx & 7;
      v8f g = {0.f, 0.f, 0.f, 0.f, 0.f, 0.f, 0.f, 0.f};
#pragma unroll
      for (int ks = 0; ks < 4; ks++) {
        v16bf a  = loadA128(sXn, mt * 16, ks * 32, CC, lane);
        v16bf bb = loadB_rm(sW1 + (nt * 4 + ks) * 512, 0, 0, 32, lane);
        g = wmma_bf16(a, bb, g);
      }
      int m0 = mt * 16 + hi * 8;
      int n = nt * 16 + (lane & 15);
      float bias1 = bb1[cc * CC + n];
#pragma unroll
      for (int r = 0; r < 8; r++) {
        float x = g[r] + bias1;
        float ge = 0.5f * x * (1.0f + erff(x * 0.70710678118654752f));
        sG[(m0 + r) * CC + n] = f2bf(ge);
      }
    }
    __syncthreads();

    // out += g @ w2_chunk
#pragma unroll
    for (int i2 = 0; i2 < 4; i2++) {
      int tidx = wv * 4 + i2;
      int mt = tidx >> 3, nt = tidx & 7;
#pragma unroll
      for (int ks = 0; ks < 4; ks++) {
        v16bf a  = loadA128(sG, mt * 16, ks * 32, CC, lane);
        v16bf bb = loadB_rm(sW2 + (ks * 8 + nt) * 512, 0, 0, 32, lane);
        acc[i2] = wmma_bf16(a, bb, acc[i2]);
      }
    }
    __syncthreads();
  }

  // epilogue: h*0.5 + x
#pragma unroll
  for (int i2 = 0; i2 < 4; i2++) {
    int tidx = wv * 4 + i2;
    int mt = tidx >> 3, nt = tidx & 7;
    int m0 = mt * 16 + hi * 8;
    int n = nt * 16 + (lane & 15);
    float bias2 = bb2[n];
#pragma unroll
    for (int r = 0; r < 8; r++) {
      long row = row0 + m0 + r;
      float h = acc[i2][r] + bias2;
      out[row * CC + n] = h * 0.5f + X[row * CC + n];
    }
  }
}

// ---------------- launcher ----------------
extern "C" void kernel_launch(void* const* d_in, const int* in_sizes, int n_in,
                              void* d_out, int out_size, void* d_ws, size_t ws_size,
                              hipStream_t stream) {
  (void)in_sizes; (void)n_in; (void)out_size; (void)ws_size;
  const float* inputs     = (const float*)d_in[0];
  const float* qkv_w      = (const float*)d_in[1];
  const float* proj_w     = (const float*)d_in[2];
  const float* proj_b     = (const float*)d_in[3];
  const float* bias_table = (const float*)d_in[4];
  const float* g1         = (const float*)d_in[5];
  const float* b1         = (const float*)d_in[6];
  const float* g2         = (const float*)d_in[7];
  const float* b2         = (const float*)d_in[8];
  const float* w1         = (const float*)d_in[9];
  const float* bb1        = (const float*)d_in[10];
  const float* w2         = (const float*)d_in[11];
  const float* bb2        = (const float*)d_in[12];
  float* out = (float*)d_out;

  char* ws = (char*)d_ws;
  unsigned short* qkvw_b = (unsigned short*)ws;                    //  98304 B
  unsigned short* pw_b   = (unsigned short*)(ws + 98304);          //  32768 B
  unsigned short* w1_b   = (unsigned short*)(ws + 131072);         // 131072 B
  unsigned short* w2_b   = (unsigned short*)(ws + 262144);         // 131072 B
  float* gBias           = (float*)(ws + 393216);                  //  65536 B
  float* X = out;  // alias residual buffer onto d_out (K3 reads-then-writes rows)

  // weight conversion + fragment swizzle + bias expansion
  k_cvt_swz<<<(CC * 3 * CC + 255) / 256, 256, 0, stream>>>(qkv_w, qkvw_b, CC, 3 * CC, 0);
  k_cvt_swz<<<(CC * CC + 255) / 256, 256, 0, stream>>>(proj_w, pw_b, CC, CC, 0);
  k_cvt_swz<<<(CC * MLPD + 255) / 256, 256, 0, stream>>>(w1, w1_b, CC, MLPD, 0);
  k_cvt_swz<<<(MLPD * CC + 255) / 256, 256, 0, stream>>>(w2, w2_b, MLPD, CC, 1);
  k_bias<<<64, 256, 0, stream>>>(bias_table, gBias);

  // fused LN1 + windowed attention + proj + reverse + residual (writes X == d_out)
  k_attn<<<NWIN, 256, 0, stream>>>(inputs, qkvw_b, pw_b, proj_b, gBias, g1, b1, X);

  // LN2 + MLP + residual (reads X, overwrites d_out)
  k_mlp<<<TOK / 64, 256, 0, stream>>>(X, g2, b2, w1_b, bb1, w2_b, bb2, out);
}
